// FFD_11295763988598
// MI455X (gfx1250) — matrix operand order repaired and checked by reading, not compiled, not run on hardware
//
#include <hip/hip_runtime.h>

typedef __attribute__((ext_vector_type(16))) __bf16 v16bf;
typedef __attribute__((ext_vector_type(8)))  __bf16 v8bf;
typedef __attribute__((ext_vector_type(8)))  float  v8f;
typedef __attribute__((ext_vector_type(4)))  int    v4i;

#define DM 256     // d_model
#define DH 1024    // d_hidden

#define GAS __attribute__((address_space(1)))
#define LAS __attribute__((address_space(3)))

// One 16-byte async global->LDS copy (tracked by ASYNCcnt, no VGPR staging)
__device__ __forceinline__ void cp16_async(const __bf16* g, __bf16* l) {
    __builtin_amdgcn_global_load_async_to_lds_b128(
        (GAS v4i*)g, (LAS v4i*)l, /*offset*/0, /*cpol*/0);
}

// Data-ready: oldest 16 async ops (current chunk) done, then WG barrier.
__device__ __forceinline__ void barrier_data_ready() {
    asm volatile("s_wait_asynccnt 16\n\t"
                 "s_barrier_signal -1\n\t"
                 "s_barrier_wait -1" ::: "memory");
}
// Compute-done: all LDS reads/writes of this wave retired, then WG barrier.
__device__ __forceinline__ void barrier_compute_done() {
    asm volatile("s_wait_dscnt 0\n\t"
                 "s_barrier_signal -1\n\t"
                 "s_barrier_wait -1" ::: "memory");
}

// ---------------------------------------------------------------------------
// Prologue: apply graph mask, split weights into bf16 hi/lo pairs in d_ws.
//   w1eff[h][d] = w1[h][d] * mask[h][d]          (fc1, row-major [H][D])
//   w2eff[d][h] = w2[d][h] * mask[h][d]          (fc2, row-major [D][H])
// ---------------------------------------------------------------------------
__global__ __launch_bounds__(256) void FFD_prep_weights(
    const float* __restrict__ w1, const float* __restrict__ w2,
    const int* __restrict__ mask,
    __bf16* __restrict__ w1hi, __bf16* __restrict__ w1lo,
    __bf16* __restrict__ w2hi, __bf16* __restrict__ w2lo)
{
    const int i = blockIdx.x * 256 + threadIdx.x;   // 0 .. H*D-1
    {
        const int h = i >> 8, d = i & 255;
        float v = w1[i] * (float)mask[h * DM + d];
        __bf16 hi = (__bf16)v;
        w1hi[i] = hi;
        w1lo[i] = (__bf16)(v - (float)hi);
    }
    {
        const int d = i >> 10, h = i & 1023;
        float v = w2[i] * (float)mask[h * DM + d];
        __bf16 hi = (__bf16)v;
        w2hi[i] = hi;
        w2lo[i] = (__bf16)(v - (float)hi);
    }
}

// ---------------------------------------------------------------------------
// Fused masked MLP. Wave owns 16 tokens; WG = 8 waves = 128 tokens.
// H streamed in 32 chunks of 32; weight slices double-buffered in LDS via
// async global->LDS copies; bf16x3 split-precision WMMA, fp32 accumulate.
// ---------------------------------------------------------------------------
__global__ __launch_bounds__(256) void FFD_main(
    const float* __restrict__ x,
    const float* __restrict__ b1,
    const float* __restrict__ b2,
    const __bf16* __restrict__ w1hi, const __bf16* __restrict__ w1lo,
    const __bf16* __restrict__ w2hi, const __bf16* __restrict__ w2lo,
    float* __restrict__ out)
{
    // [buf][array: 0=W1hi [32][256], 1=W1lo, 2=W2hi [256][32], 3=W2lo]
    __shared__ __attribute__((aligned(32))) __bf16 sW[2][4][32 * DM];
    // Per-wave h staging tile [16 rows][32 cols], C-layout -> A-layout
    __shared__ __attribute__((aligned(32))) __bf16 sHhi[8][16 * 32];
    __shared__ __attribute__((aligned(32))) __bf16 sHlo[8][16 * 32];

    const int tid   = threadIdx.x;
    const int lane  = tid & 31;
    const int wave  = tid >> 5;
    const int lrow  = lane & 15;
    const int lhalf = lane >> 4;

    const int m0 = (blockIdx.x * 8 + wave) * 16;

    // Issue the 16 async b128 copies of one 64KB weight slice (h0 = chunk*32)
    auto prefetch = [&](int buf, int h0) {
#pragma unroll
        for (int r = 0; r < 4; ++r) {
            const int o = (tid + 256 * r) * 8;                 // bf16 elements
            cp16_async(w1hi + (size_t)h0 * DM + o, &sW[buf][0][o]);
            cp16_async(w1lo + (size_t)h0 * DM + o, &sW[buf][1][o]);
        }
#pragma unroll
        for (int r = 0; r < 4; ++r) {                          // row d = tid
            cp16_async(w2hi + (size_t)tid * DH + h0 + r * 8, &sW[buf][2][tid * 32 + r * 8]);
            cp16_async(w2lo + (size_t)tid * DH + h0 + r * 8, &sW[buf][3][tid * 32 + r * 8]);
        }
    };

    // ---- preload x as bf16 hi/lo A-fragments: 8 K-chunks of 32 over D ----
    // A k order used here (contiguous, B matches): lane-half 0 holds K {0..15},
    //                          lane-half 1 holds K {16..31}
    v16bf xhi[8], xlo[8];
    {
        const float* xrow = x + (size_t)(m0 + lrow) * DM;
#pragma unroll
        for (int kc = 0; kc < 8; ++kc) {
            const float* pb = xrow + kc * 32;
#pragma unroll
            for (int e = 0; e < 8; ++e) {
                float f0 = pb[lhalf * 16 + e];
                float f1 = pb[lhalf * 16 + 8 + e];
                __bf16 h0 = (__bf16)f0;
                __bf16 h1 = (__bf16)f1;
                xhi[kc][e]     = h0;
                xhi[kc][8 + e] = h1;
                xlo[kc][e]     = (__bf16)(f0 - (float)h0);
                xlo[kc][8 + e] = (__bf16)(f1 - (float)h1);
            }
        }
    }

    v8f yacc[16];
#pragma unroll
    for (int n = 0; n < 16; ++n) yacc[n] = (v8f){};

    prefetch(0, 0);   // prime the pipeline

    for (int hc = 0; hc < 32; ++hc) {
        const int cur = hc & 1;
        const int h0  = hc * 32;
        // issue next chunk into the other buffer (wraps harmlessly at end)
        prefetch(cur ^ 1, ((hc + 1) & 31) * 32);
        barrier_data_ready();   // s_wait_asynccnt 16 -> chunk hc has landed

        // ---- fc1: h[16x32] = x[16x256] * W1eff[32x256]^T (bf16x3) ----
        // 3 independent accumulator chains per tile to hide WMMA RAW hazard
#pragma unroll
        for (int t = 0; t < 2; ++t) {
            v8f chh = (v8f){}, chl = (v8f){}, clh = (v8f){};
            const __bf16* wb  = &sW[cur][0][(t * 16 + lrow) * DM + lhalf * 16];
            const __bf16* wbl = &sW[cur][1][(t * 16 + lrow) * DM + lhalf * 16];
#pragma unroll
            for (int kc = 0; kc < 8; ++kc) {
                v16bf bhi = *(const v16bf*)(wb  + kc * 32);
                v16bf blo = *(const v16bf*)(wbl + kc * 32);
                chh = __builtin_amdgcn_wmma_f32_16x16x32_bf16(false, xhi[kc], false, bhi, (short)0, chh, false, false);
                chl = __builtin_amdgcn_wmma_f32_16x16x32_bf16(false, xhi[kc], false, blo, (short)0, chl, false, false);
                clh = __builtin_amdgcn_wmma_f32_16x16x32_bf16(false, xlo[kc], false, bhi, (short)0, clh, false, false);
            }
            v8f c = chh + chl + clh;
            // bias + ReLU, hi/lo split, stage (C layout: M = v+8*lhalf, N = lrow)
            const float bias = b1[h0 + t * 16 + lrow];
#pragma unroll
            for (int v = 0; v < 8; ++v) {
                float hv = c[v] + bias;
                hv = hv > 0.0f ? hv : 0.0f;
                __bf16 hi = (__bf16)hv;
                const int row = v + 8 * lhalf;
                const int col = t * 16 + lrow;
                sHhi[wave][row * 32 + col] = hi;
                sHlo[wave][row * 32 + col] = (__bf16)(hv - (float)hi);
            }
        }

        // ---- fc2: y[16x256] += h[16x32] * W2eff[256x32]^T (bf16x3) ----
        // same-wave LDS store->load: DS ops execute in order within a wave
        const __bf16* hb = &sHhi[wave][lrow * 32];
        const __bf16* lb = &sHlo[wave][lrow * 32];
        v8bf a0 = *(const v8bf*)(hb + lhalf * 16);
        v8bf a1 = *(const v8bf*)(hb + lhalf * 16 + 8);
        v16bf ahi = __builtin_shufflevector(a0, a1, 0,1,2,3,4,5,6,7,8,9,10,11,12,13,14,15);
        a0 = *(const v8bf*)(lb + lhalf * 16);
        a1 = *(const v8bf*)(lb + lhalf * 16 + 8);
        v16bf alo = __builtin_shufflevector(a0, a1, 0,1,2,3,4,5,6,7,8,9,10,11,12,13,14,15);

#pragma unroll
        for (int n = 0; n < 16; ++n) {
            const __bf16* p  = &sW[cur][2][(n * 16 + lrow) * 32 + lhalf * 16];
            const __bf16* pl = &sW[cur][3][(n * 16 + lrow) * 32 + lhalf * 16];
            v16bf bhi = *(const v16bf*)p;
            v16bf blo = *(const v16bf*)pl;
            yacc[n] = __builtin_amdgcn_wmma_f32_16x16x32_bf16(false, ahi, false, bhi, (short)0, yacc[n], false, false);
            yacc[n] = __builtin_amdgcn_wmma_f32_16x16x32_bf16(false, ahi, false, blo, (short)0, yacc[n], false, false);
            yacc[n] = __builtin_amdgcn_wmma_f32_16x16x32_bf16(false, alo, false, bhi, (short)0, yacc[n], false, false);
        }
        barrier_compute_done();   // protect buffers before they are rewritten
    }

    // ---- epilogue: add b2, store y ----
#pragma unroll
    for (int n = 0; n < 16; ++n) {
        const float bias = b2[n * 16 + lrow];
#pragma unroll
        for (int v = 0; v < 8; ++v) {
            const int row = m0 + v + 8 * lhalf;
            out[(size_t)row * DM + n * 16 + lrow] = yacc[n][v] + bias;
        }
    }
}

// ---------------------------------------------------------------------------
extern "C" void kernel_launch(void* const* d_in, const int* in_sizes, int n_in,
                              void* d_out, int out_size, void* d_ws, size_t ws_size,
                              hipStream_t stream) {
    const float* x    = (const float*)d_in[0];
    const float* w1   = (const float*)d_in[1];
    const float* b1   = (const float*)d_in[2];
    const float* w2   = (const float*)d_in[3];
    const float* b2   = (const float*)d_in[4];
    const int*   mask = (const int*)d_in[5];
    float* out = (float*)d_out;

    char* ws = (char*)d_ws;
    __bf16* w1hi = (__bf16*)(ws);                       // 512 KB each
    __bf16* w1lo = (__bf16*)(ws +  512 * 1024);
    __bf16* w2hi = (__bf16*)(ws + 1024 * 1024);
    __bf16* w2lo = (__bf16*)(ws + 1536 * 1024);

    FFD_prep_weights<<<1024, 256, 0, stream>>>(w1, w2, mask, w1hi, w1lo, w2hi, w2lo);
    FFD_main<<<1024, 256, 0, stream>>>(x, b1, b2, w1hi, w1lo, w2hi, w2lo, out);
}
